// EncodeProcessDecode_24945170055802
// MI455X (gfx1250) — matrix operand order repaired and checked by reading, not compiled, not run on hardware
//
#include <hip/hip_runtime.h>
#include <hip/hip_bf16.h>

// ---------------------------------------------------------------------------
// EncodeProcessDecode GNN for MI455X (gfx1250), wave32 + WMMA bf16.
// All GEMMs use v_wmma_f32_16x16x32_bf16 (fp32 accumulate). Activations kept
// as f32 master copies (residual/LN) with bf16 mirrors for GEMM operands.
// A-tiles are staged into LDS with GLOBAL_LOAD_ASYNC_TO_LDS_B128 when the
// toolchain exposes the builtin (ASYNCcnt path), else b128 copies via VGPRs.
// ---------------------------------------------------------------------------

typedef __bf16 bf16_t;
typedef __attribute__((ext_vector_type(16))) __bf16 v16bf;
typedef __attribute__((ext_vector_type(8)))  __bf16 v8bf;
typedef __attribute__((ext_vector_type(8)))  float    v8f;
typedef __attribute__((ext_vector_type(4)))  unsigned int u32x4;
typedef __attribute__((ext_vector_type(4)))  int i32x4;

// Address-space qualified int4 pointers for the async-to-LDS builtin:
// clang's HIP printer shows AS1 as "__device__", AS3 as "__shared__".
typedef __attribute__((address_space(1))) i32x4 as1_i32x4;
typedef __attribute__((address_space(3))) i32x4 as3_i32x4;

#define LN_EPS 1e-5f

#if defined(__has_builtin)
#  if __has_builtin(__builtin_amdgcn_global_load_async_to_lds_b128)
#    define HAVE_ASYNC_LDS 1
#  else
#    define HAVE_ASYNC_LDS 0
#  endif
#else
#  define HAVE_ASYNC_LDS 0
#endif

union AB16 { v16bf v; v8bf h[2]; };

__device__ __forceinline__ void wmma_acc(v8f& acc, const v16bf& a, const v16bf& b) {
  // 8 args: (neg_a, A, neg_b, B, c_mod, C, reuse_a, reuse_b)
  acc = __builtin_amdgcn_wmma_f32_16x16x32_bf16(false, a, false, b, (short)0, acc,
                                                false, false);
}

// Stage 16 bytes global -> LDS (async direct-to-LDS when available).
__device__ __forceinline__ void stage16(bf16_t* lds_dst, const bf16_t* gsrc) {
#if HAVE_ASYNC_LDS
  __builtin_amdgcn_global_load_async_to_lds_b128(
      (as1_i32x4*)(void*)gsrc, (as3_i32x4*)(void*)lds_dst, 0, 0);
#else
  *(u32x4*)lds_dst = *(const u32x4*)gsrc;
#endif
}

__device__ __forceinline__ void zero16(bf16_t* lds_dst) {
  u32x4 z = {0u, 0u, 0u, 0u};
  *(u32x4*)lds_dst = z;
}

// Wait for this wave's outstanding async LDS fills (no-op on fallback path).
__device__ __forceinline__ void stage_wait() {
#if HAVE_ASYNC_LDS
#  if __has_builtin(__builtin_amdgcn_s_wait_asynccnt)
  __builtin_amdgcn_s_wait_asynccnt(0);
#  else
  asm volatile("s_wait_asynccnt 0x0" ::: "memory");
#  endif
#endif
}

// ---------------------------------------------------------------------------
// Shared compute+store: A tile (64 x KPAD bf16) staged in LDS, B from packed
// global weights laid out per WMMA B-fragment: [kt][ntile][lane32][16].
// Wave w owns output n-tile w. 4 M-subtiles of 16 rows each.
// ---------------------------------------------------------------------------
template<int KT, bool RELU, bool OUT_BF16, int NT>
__device__ __forceinline__ void gemm_tile_compute(
    const bf16_t* As, int stride,
    const bf16_t* __restrict__ Wp, const float* __restrict__ bias,
    void* __restrict__ Out, int M, int row0, int NOUT, int NSTORE)
{
  const int tid  = threadIdx.x;
  const int wave = tid >> 5;
  const int lane = tid & 31;
  if (wave >= NT) return;                 // uniform per-wave branch: EXEC stays all-1s
  const int lhalf = lane >> 4;            // 0 / 1
  const int l16   = lane & 15;

  v8f zero = {0.f,0.f,0.f,0.f,0.f,0.f,0.f,0.f};
  v8f acc[4];
#pragma unroll
  for (int i = 0; i < 4; ++i) acc[i] = zero;

  for (int kt = 0; kt < KT; ++kt) {
    // B fragment: 16 contiguous bf16 per lane (pre-swizzled) -> two b128 loads
    AB16 b;
    const v8bf* wp = (const v8bf*)(Wp + (((size_t)kt * NT + wave) * 32 + lane) * 16);
    b.h[0] = wp[0];
    b.h[1] = wp[1];
#pragma unroll
    for (int mt = 0; mt < 4; ++mt) {
      // A fragment per ISA layout: lanes<16 hold K {0..7,16..23}, lanes>=16 {8..15,24..31}
      const bf16_t* ap = As + (mt * 16 + l16) * stride + kt * 32 + lhalf * 8;
      AB16 a;
      a.h[0] = *(const v8bf*)(ap);
      a.h[1] = *(const v8bf*)(ap + 16);
      wmma_acc(acc[mt], a.v, b.v);
    }
  }

  const int nbase = wave * 16 + l16;
  if (nbase < NSTORE) {
    const float bsc = bias[nbase];
#pragma unroll
    for (int mt = 0; mt < 4; ++mt) {
#pragma unroll
      for (int r = 0; r < 8; ++r) {
        // C/D layout: VGPR r -> M = r + 8*(lane>=16), N = lane&15
        int m = row0 + mt * 16 + lhalf * 8 + r;
        if (m < M) {
          float v = acc[mt][r] + bsc;
          if (RELU) v = fmaxf(v, 0.f);
          if (OUT_BF16) ((bf16_t*)Out)[(size_t)m * NOUT + nbase] = (bf16_t)v;
          else          ((float*)Out)[(size_t)m * NOUT + nbase] = v;
        }
      }
    }
  }
}

// ---------------------------------------------------------------------------
// Generic MLP layer: A is a dense [M x KR] bf16 matrix. KR padded to KT*32.
// ---------------------------------------------------------------------------
template<int KR, int KT, bool RELU, bool OUT_BF16, int NT>
__global__ __launch_bounds__(256) void mlp_layer_kernel(
    const bf16_t* __restrict__ A, const bf16_t* __restrict__ Wp,
    const float* __restrict__ bias, void* __restrict__ Out,
    int M, int NOUT, int NSTORE)
{
  constexpr int KPAD   = KT * 32;
  constexpr int STRIDE = KPAD + 8;        // +16B padding breaks LDS bank conflicts
  __shared__ bf16_t As[64 * STRIDE];
  const int row0 = blockIdx.x * 64;
  const int tid  = threadIdx.x;

  constexpr int CPR = KR / 8;             // 16B chunks per source row
  for (int idx = tid; idx < 64 * CPR; idx += 256) {
    int row = idx / CPR, c = idx - row * CPR;
    int g = row0 + row;
    bf16_t* dst = &As[row * STRIDE + c * 8];
    if (g < M) stage16(dst, A + (size_t)g * KR + c * 8);
    else       zero16(dst);
  }
  if constexpr (KPAD > KR) {               // zero-fill K padding
    constexpr int PC = (KPAD - KR) / 8;
    for (int idx = tid; idx < 64 * PC; idx += 256) {
      int row = idx / PC, c = idx - row * PC;
      zero16(&As[row * STRIDE + KR + c * 8]);
    }
  }
  stage_wait();
  __syncthreads();
  gemm_tile_compute<KT, RELU, OUT_BF16, NT>(As, STRIDE, Wp, bias, Out, M, row0, NOUT, NSTORE);
}

// ---------------------------------------------------------------------------
// Edge-MLP first layer, gather fused: A row = concat(x[s], x[r], e), K = 384.
// ---------------------------------------------------------------------------
__global__ __launch_bounds__(256) void edge_input_kernel(
    const bf16_t* __restrict__ xb, const bf16_t* __restrict__ eb,
    const int* __restrict__ senders, const int* __restrict__ receivers,
    const bf16_t* __restrict__ Wp, const float* __restrict__ bias,
    bf16_t* __restrict__ Out, int E)
{
  constexpr int STRIDE = 384 + 8;
  __shared__ bf16_t As[64 * STRIDE];
  const int row0 = blockIdx.x * 64;
  const int tid  = threadIdx.x;

  for (int idx = tid; idx < 64 * 48; idx += 256) {  // 48 chunks of 16B = 384 bf16
    int row = idx / 48, c = idx - row * 48;
    int g = row0 + row;
    bf16_t* dst = &As[row * STRIDE + c * 8];
    if (g < E) {
      int seg = c >> 4, off = c & 15;
      const bf16_t* src;
      if (seg == 0)      src = xb + (size_t)senders[g]   * 128;
      else if (seg == 1) src = xb + (size_t)receivers[g] * 128;
      else               src = eb + (size_t)g            * 128;
      stage16(dst, src + off * 8);
    } else {
      zero16(dst);
    }
  }
  stage_wait();
  __syncthreads();
  gemm_tile_compute<12, true, true, 8>(As, STRIDE, Wp, bias, (void*)Out, E, row0, 128, 128);
}

// ---------------------------------------------------------------------------
// Node-MLP first layer: A row = concat(x, agg), K = 256.
// ---------------------------------------------------------------------------
__global__ __launch_bounds__(256) void node_input_kernel(
    const bf16_t* __restrict__ xb, const bf16_t* __restrict__ aggb,
    const bf16_t* __restrict__ Wp, const float* __restrict__ bias,
    bf16_t* __restrict__ Out, int N)
{
  constexpr int STRIDE = 256 + 8;
  __shared__ bf16_t As[64 * STRIDE];
  const int row0 = blockIdx.x * 64;
  const int tid  = threadIdx.x;

  for (int idx = tid; idx < 64 * 32; idx += 256) {
    int row = idx / 32, c = idx - row * 32;
    int g = row0 + row;
    bf16_t* dst = &As[row * STRIDE + c * 8];
    if (g < N) {
      int seg = c >> 4, off = c & 15;
      const bf16_t* src = (seg == 0) ? xb + (size_t)g * 128 : aggb + (size_t)g * 128;
      stage16(dst, src + off * 8);
    } else {
      zero16(dst);
    }
  }
  stage_wait();
  __syncthreads();
  gemm_tile_compute<8, true, true, 8>(As, STRIDE, Wp, bias, (void*)Out, N, row0, 128, 128);
}

// ---------------------------------------------------------------------------
// Pack f32 [K x Nsrc] weights into WMMA B-fragment order: [kt][ntile][lane][16]
// with zero padding for k >= Kreal or n >= Nsrc.
// ---------------------------------------------------------------------------
__global__ __launch_bounds__(256) void pack_weight_kernel(
    const float* __restrict__ w, bf16_t* __restrict__ dst,
    int Kreal, int Nsrc, int KT, int NT)
{
  long long total = (long long)KT * NT * 512;
  long long idx = (long long)blockIdx.x * 256 + threadIdx.x;
  if (idx >= total) return;
  int kt   = (int)(idx / ((long long)NT * 512));
  int rem  = (int)(idx - (long long)kt * NT * 512);
  int nt   = rem / 512;
  int q    = rem - nt * 512;
  int lane = q >> 4;
  int i    = q & 15;
  int n = nt * 16 + (lane & 15);
  int k = kt * 32 + (lane >> 4) * 8 + (i & 7) + (i >> 3) * 16;   // B layout: lanes<16 K {0..7,16..23}, lanes>=16 K {8..15,24..31}
  float v = (k < Kreal && n < Nsrc) ? w[(size_t)k * Nsrc + n] : 0.f;
  dst[idx] = (bf16_t)v;
}

// ---------------------------------------------------------------------------
// LayerNorm (+optional residual add) over 128 features; one row per block.
// Writes f32 master (resid), bf16 mirror (outb), optional raw LN value (newout).
// ---------------------------------------------------------------------------
__global__ __launch_bounds__(128) void ln_kernel(
    const float* __restrict__ h, const float* __restrict__ g, const float* __restrict__ be,
    float* __restrict__ resid, bf16_t* __restrict__ outb, float* __restrict__ newout,
    int add)
{
  const int row = blockIdx.x;
  const int t   = threadIdx.x;
  __shared__ float red[128];
  const size_t i = (size_t)row * 128 + t;
  float v = h[i];
  red[t] = v; __syncthreads();
  for (int s = 64; s > 0; s >>= 1) { if (t < s) red[t] += red[t + s]; __syncthreads(); }
  float mu = red[0] * (1.0f / 128.0f);
  __syncthreads();
  float d = v - mu;
  red[t] = d * d; __syncthreads();
  for (int s = 64; s > 0; s >>= 1) { if (t < s) red[t] += red[t + s]; __syncthreads(); }
  float var = red[0] * (1.0f / 128.0f);
  float ln = d * rsqrtf(var + LN_EPS) * g[t] + be[t];
  if (newout) newout[i] = ln;
  float rv = add ? (resid[i] + ln) : ln;
  resid[i] = rv;
  outb[i]  = (bf16_t)rv;
}

__global__ __launch_bounds__(256) void scatter_add_kernel(
    const float* __restrict__ enew, const int* __restrict__ recv,
    float* __restrict__ agg, long long total)
{
  long long i = (long long)blockIdx.x * 256 + threadIdx.x;
  if (i >= total) return;
  int e = (int)(i >> 7), c = (int)(i & 127);
  unsafeAtomicAdd(&agg[(size_t)recv[e] * 128 + c], enew[i]);  // global_atomic_add_f32
}

__global__ __launch_bounds__(256) void f32_to_bf16_kernel(
    const float* __restrict__ in, bf16_t* __restrict__ out, long long n)
{
  long long i = (long long)blockIdx.x * 256 + threadIdx.x;
  if (i < n) out[i] = (bf16_t)in[i];
}

__global__ __launch_bounds__(256) void zero_f32_kernel(float* __restrict__ p, long long n)
{
  long long i = (long long)blockIdx.x * 256 + threadIdx.x;
  if (i < n) p[i] = 0.f;
}

// ---------------------------------------------------------------------------
// Host orchestration
// ---------------------------------------------------------------------------
extern "C" void kernel_launch(void* const* d_in, const int* in_sizes, int n_in,
                              void* d_out, int out_size, void* d_ws, size_t ws_size,
                              hipStream_t stream)
{
  (void)n_in; (void)out_size; (void)ws_size;
  const int N = in_sizes[0] / 16;   // 20000
  const int E = in_sizes[2];        // 160000
  const int* senders   = (const int*)d_in[2];
  const int* receivers = (const int*)d_in[3];
  auto F = [&](int i) { return (const float*)d_in[i]; };

  // --- weight descriptors (d_in index, Kreal, KT, NT, Nsrc), insertion order ---
  struct WD { int idx, K, KT, NT, Nsrc; };
  WD wds[33]; int nw = 0;
  auto add_mlp3 = [&](int i0, int K0, int KT0) {
    wds[nw++] = {i0 + 0, K0, KT0, 8, 128};
    wds[nw++] = {i0 + 2, 128, 4, 8, 128};
    wds[nw++] = {i0 + 4, 128, 4, 8, 128};
  };
  add_mlp3(4, 16, 1);                 // enc_node [16,128,128,128]
  add_mlp3(12, 8, 1);                 // enc_edge [8,128,128,128]
  for (int s = 0; s < 4; ++s) {       // mp steps: edge [384,...], node [256,...]
    int b = 20 + 16 * s;
    add_mlp3(b, 384, 12);
    add_mlp3(b + 8, 256, 8);
  }
  wds[nw++] = {84, 128, 4, 8, 128};   // dec w0
  wds[nw++] = {86, 128, 4, 8, 128};   // dec w1
  wds[nw++] = {88, 128, 4, 1, 3};     // dec w2 (128x3, N padded to one tile)

  size_t woff[33], wtot = 0;
  for (int i = 0; i < 33; ++i) { woff[i] = wtot; wtot += (size_t)wds[i].KT * wds[i].NT * 512; }

  // --- workspace carve ---
  char* base = (char*)d_ws; size_t off = 0;
  auto carve = [&](size_t bytes) -> void* {
    void* r = base + off; off += (bytes + 255) & ~(size_t)255; return r;
  };
  bf16_t* wpk  = (bf16_t*)carve(wtot * 2);
  bf16_t* nfb  = (bf16_t*)carve((size_t)N * 16 * 2);
  bf16_t* efb  = (bf16_t*)carve((size_t)E * 8 * 2);
  float*  xf   = (float*) carve((size_t)N * 128 * 4);   // node latent f32
  bf16_t* xb   = (bf16_t*)carve((size_t)N * 128 * 2);   // node latent bf16
  float*  el   = (float*) carve((size_t)E * 128 * 4);   // edge latent f32
  bf16_t* eb   = (bf16_t*)carve((size_t)E * 128 * 2);   // edge latent bf16
  float*  enew = (float*) carve((size_t)E * 128 * 4);
  float*  agg  = (float*) carve((size_t)N * 128 * 4);
  bf16_t* aggb = (bf16_t*)carve((size_t)N * 128 * 2);
  bf16_t* hb0  = (bf16_t*)carve((size_t)E * 128 * 2);   // hidden ping
  bf16_t* hb1  = (bf16_t*)carve((size_t)E * 128 * 2);   // hidden pong
  float*  hf   = (float*) carve((size_t)E * 128 * 4);   // pre-LN f32

  // --- pack weights to bf16 WMMA fragment order ---
  for (int i = 0; i < 33; ++i) {
    long long tot = (long long)wds[i].KT * wds[i].NT * 512;
    pack_weight_kernel<<<(int)((tot + 255) / 256), 256, 0, stream>>>(
        F(wds[i].idx), wpk + woff[i], wds[i].K, wds[i].Nsrc, wds[i].KT, wds[i].NT);
  }
  // --- convert raw inputs to bf16 ---
  f32_to_bf16_kernel<<<(int)(((long long)N * 16 + 255) / 256), 256, 0, stream>>>(F(0), nfb, (long long)N * 16);
  f32_to_bf16_kernel<<<(int)(((long long)E * 8 + 255) / 256), 256, 0, stream>>>(F(1), efb, (long long)E * 8);

  const int gN = (N + 63) / 64, gE = (E + 63) / 64;
  const int cN = (int)(((long long)N * 128 + 255) / 256);

  // --- encoders ---
  mlp_layer_kernel<16, 1, true,  true,  8><<<gN, 256, 0, stream>>>(nfb, wpk + woff[0], F(5), (void*)hb0, N, 128, 128);
  mlp_layer_kernel<128,4, true,  true,  8><<<gN, 256, 0, stream>>>(hb0, wpk + woff[1], F(7), (void*)hb1, N, 128, 128);
  mlp_layer_kernel<128,4, false, false, 8><<<gN, 256, 0, stream>>>(hb1, wpk + woff[2], F(9), (void*)hf,  N, 128, 128);
  ln_kernel<<<N, 128, 0, stream>>>(hf, F(10), F(11), xf, xb, nullptr, 0);

  mlp_layer_kernel<8,  1, true,  true,  8><<<gE, 256, 0, stream>>>(efb, wpk + woff[3], F(13), (void*)hb0, E, 128, 128);
  mlp_layer_kernel<128,4, true,  true,  8><<<gE, 256, 0, stream>>>(hb0, wpk + woff[4], F(15), (void*)hb1, E, 128, 128);
  mlp_layer_kernel<128,4, false, false, 8><<<gE, 256, 0, stream>>>(hb1, wpk + woff[5], F(17), (void*)hf,  E, 128, 128);
  ln_kernel<<<E, 128, 0, stream>>>(hf, F(18), F(19), el, eb, nullptr, 0);

  // --- message passing ---
  for (int s = 0; s < 4; ++s) {
    int b = 20 + 16 * s, w = 6 + 6 * s;
    // edge MLP (gather fused into first layer)
    edge_input_kernel<<<gE, 256, 0, stream>>>(xb, eb, senders, receivers, wpk + woff[w], F(b + 1), hb0, E);
    mlp_layer_kernel<128,4, true,  true,  8><<<gE, 256, 0, stream>>>(hb0, wpk + woff[w + 1], F(b + 3), (void*)hb1, E, 128, 128);
    mlp_layer_kernel<128,4, false, false, 8><<<gE, 256, 0, stream>>>(hb1, wpk + woff[w + 2], F(b + 5), (void*)hf,  E, 128, 128);
    ln_kernel<<<E, 128, 0, stream>>>(hf, F(b + 6), F(b + 7), el, eb, enew, 1);
    // aggregate
    zero_f32_kernel<<<cN, 256, 0, stream>>>(agg, (long long)N * 128);
    scatter_add_kernel<<<(int)(((long long)E * 128 + 255) / 256), 256, 0, stream>>>(enew, receivers, agg, (long long)E * 128);
    f32_to_bf16_kernel<<<cN, 256, 0, stream>>>(agg, aggb, (long long)N * 128);
    // node MLP
    node_input_kernel<<<gN, 256, 0, stream>>>(xb, aggb, wpk + woff[w + 3], F(b + 9), hb0, N);
    mlp_layer_kernel<128,4, true,  true,  8><<<gN, 256, 0, stream>>>(hb0, wpk + woff[w + 4], F(b + 11), (void*)hb1, N, 128, 128);
    mlp_layer_kernel<128,4, false, false, 8><<<gN, 256, 0, stream>>>(hb1, wpk + woff[w + 5], F(b + 13), (void*)hf,  N, 128, 128);
    ln_kernel<<<N, 128, 0, stream>>>(hf, F(b + 14), F(b + 15), xf, xb, nullptr, 1);
  }

  // --- decoder ---
  mlp_layer_kernel<128,4, true,  true,  8><<<gN, 256, 0, stream>>>(xb,  wpk + woff[30], F(85), (void*)hb0, N, 128, 128);
  mlp_layer_kernel<128,4, true,  true,  8><<<gN, 256, 0, stream>>>(hb0, wpk + woff[31], F(87), (void*)hb1, N, 128, 128);
  mlp_layer_kernel<128,4, false, false, 1><<<gN, 256, 0, stream>>>(hb1, wpk + woff[32], F(89), d_out, N, 3, 3);
}
